// Qwen3MoeAttention_1090921693842
// MI455X (gfx1250) — matrix operand order repaired and checked by reading, not compiled, not run on hardware
//
#include <hip/hip_runtime.h>

// ---------------------------------------------------------------------------
// Types for CDNA5 WMMA (wave32)
// ---------------------------------------------------------------------------
typedef __attribute__((ext_vector_type(16))) __bf16       v16bf;
typedef __attribute__((ext_vector_type(8)))  float        v8f;
typedef __attribute__((ext_vector_type(4)))  unsigned int uint4v;

__device__ __forceinline__ unsigned short f2bf(float f) {
  union { float f; unsigned int u; } v; v.f = f;
  unsigned int u = v.u;
  u += 0x7FFFu + ((u >> 16) & 1u);          // round-to-nearest-even
  return (unsigned short)(u >> 16);
}
__device__ __forceinline__ float bf2f(unsigned short b) {
  union { unsigned int u; float f; } v; v.u = ((unsigned int)b) << 16;
  return v.f;
}

// LDS byte offset of a generic pointer to __shared__ (aperture in high bits,
// offset in addr[31:0] per CDNA5 ISA 10.2)
__device__ __forceinline__ unsigned lds_off(const void* p) {
  return (unsigned)(unsigned long long)(size_t)p;
}

// Async global -> LDS copy of 16 bytes (CDNA5 GLOBAL_LOAD_ASYNC_TO_LDS_B128,
// tracked by ASYNCcnt). GV addressing: 64-bit VGPR address, LDS addr in VGPR.
__device__ __forceinline__ void async_load_b128(unsigned lds_addr,
                                                const void* gptr) {
  asm volatile("global_load_async_to_lds_b128 %0, %1, off"
               :: "v"(lds_addr), "v"((unsigned long long)(size_t)gptr)
               : "memory");
}
__device__ __forceinline__ void wait_async0() {
  asm volatile("s_wait_asynccnt 0x0" ::: "memory");
}
__device__ __forceinline__ void wait_ds0() {
  asm volatile("s_wait_dscnt 0x0" ::: "memory");
}

// Build a v16bf fragment from two 16-byte chunks
__device__ __forceinline__ v16bf frag2x(const unsigned short* p0,
                                        const unsigned short* p1) {
  union { uint4v q[2]; v16bf f; } u;
  u.q[0] = *(const uint4v*)p0;
  u.q[1] = *(const uint4v*)p1;
  return u.f;
}

// A-fragment: 16x32 bf16, row-major source, leading dim ld (elements).
// ISA layout: lane(0-15) row M=lane holds K = half*8+e (e<8), 16+half*8+e-8
__device__ __forceinline__ v16bf load_frag_a(const unsigned short* base, int ld,
                                             int lane) {
  int half = lane >> 4, r = lane & 15;
  const unsigned short* p = base + r * ld + half * 8;
  return frag2x(p, p + 16);
}

// B-fragment: 32x16 bf16 read from K-major (transposed) storage src[n][k],
// leading dim ldk. lane holds col N=lane&15, K = half*8+e (e<8), 16+half*8+e-8
__device__ __forceinline__ v16bf load_frag_bt(const unsigned short* base,
                                              int ldk, int lane) {
  int half = lane >> 4, n = lane & 15;
  const unsigned short* p = base + n * ldk + half * 8;
  return frag2x(p, p + 16);
}

// ---------------------------------------------------------------------------
// f32 -> bf16 conversion (grid-stride)
// ---------------------------------------------------------------------------
__global__ void cvt_f32_bf16(const float* __restrict__ in,
                             unsigned short* __restrict__ out, int n) {
  for (int i = blockIdx.x * blockDim.x + threadIdx.x; i < n;
       i += gridDim.x * blockDim.x)
    out[i] = f2bf(in[i]);
}

// ---------------------------------------------------------------------------
// bf16 GEMM: C[MxN] = A[MxK] * B[KxN], f32 accumulate via WMMA.
// Block = 256 threads (8 waves), 128x128 tile, K-step 32, double-buffered LDS.
// A tiles arrive via async global->LDS; B tiles are transposed on store.
// Per wave per K-step: 2 A-frags x 4 B-frags = 8 WMMAs.
// ---------------------------------------------------------------------------
template <bool BF16OUT>
__global__ __launch_bounds__(256)
void gemm_bf16_wmma(const unsigned short* __restrict__ A,
                    const unsigned short* __restrict__ Bm,
                    void* __restrict__ Cout, int M, int N, int Kd) {
  __shared__ unsigned short sA[2][128 * 32];   // [m][k]
  __shared__ unsigned short sB[2][128 * 32];   // transposed: [n][k]
  const int tid = threadIdx.x;
  const int lane = tid & 31;
  const int wave = tid >> 5;
  const int wm = (wave & 3) * 32;   // wave row offset (2 A-frags)
  const int wn = (wave >> 2) * 64;  // wave col offset (4 B-frags)
  const int m0 = blockIdx.y * 128;
  const int n0 = blockIdx.x * 128;
  const unsigned sA_base = lds_off(&sA[0][0]);

  v8f acc[2][4];
#pragma unroll
  for (int a = 0; a < 2; ++a)
#pragma unroll
    for (int t = 0; t < 4; ++t)
#pragma unroll
      for (int e = 0; e < 8; ++e) acc[a][t][e] = 0.0f;

  // A tile fill: 128x32 bf16 = 512 x 16B chunks, 2 per thread, async
  auto fillA = [&](int bsel, int kk) {
    unsigned base = sA_base + (unsigned)(bsel * 128 * 32 * 2);
    int chunk = tid;
#pragma unroll
    for (int it = 0; it < 2; ++it, chunk += 256) {
      int r = chunk >> 2;
      int c = (chunk & 3) * 8;
      async_load_b128(base + (unsigned)((r * 32 + c) * 2),
                      A + (size_t)(m0 + r) * Kd + kk + c);
    }
  };
  // B tile fill: 32x128 row-major global, stored transposed [n][k]
  auto fillB = [&](int bsel, int kk) {
    int chunk = tid;
#pragma unroll
    for (int it = 0; it < 2; ++it, chunk += 256) {
      int kr = chunk >> 4;           // 0..31
      int nc = (chunk & 15) * 8;     // 0..120
      uint4v d = *(const uint4v*)(Bm + (size_t)(kk + kr) * N + n0 + nc);
      const unsigned short* ds = (const unsigned short*)&d;
#pragma unroll
      for (int j = 0; j < 8; ++j) sB[bsel][(nc + j) * 32 + kr] = ds[j];
    }
  };

  fillA(0, 0);
  fillB(0, 0);

  int buf = 0;
  for (int k0 = 0; k0 < Kd; k0 += 32, buf ^= 1) {
    wait_async0();          // previous fill of `buf` complete (this wave)
    __syncthreads();        // all waves' fills of `buf` visible
    int k1 = k0 + 32;
    if (k1 < Kd) {          // overlap: fill other buffer while computing
      fillA(buf ^ 1, k1);
      fillB(buf ^ 1, k1);
    }
    const unsigned short* pa = sA[buf] + wm * 32;
    v16bf af0 = load_frag_a(pa, 32, lane);
    v16bf af1 = load_frag_a(pa + 16 * 32, 32, lane);
#pragma unroll
    for (int t = 0; t < 4; ++t) {
      v16bf bf = load_frag_bt(sB[buf] + (wn + t * 16) * 32, 32, lane);
      acc[0][t] = __builtin_amdgcn_wmma_f32_16x16x32_bf16(
          false, af0, false, bf, (short)0, acc[0][t], false, false);
      acc[1][t] = __builtin_amdgcn_wmma_f32_16x16x32_bf16(
          false, af1, false, bf, (short)0, acc[1][t], false, false);
    }
  }

  const int half = lane >> 4, nl = lane & 15;
#pragma unroll
  for (int a = 0; a < 2; ++a)
#pragma unroll
    for (int t = 0; t < 4; ++t)
#pragma unroll
      for (int e = 0; e < 8; ++e) {
        int row = m0 + wm + a * 16 + e + 8 * half;
        int col = n0 + wn + t * 16 + nl;
        if (BF16OUT)
          ((unsigned short*)Cout)[(size_t)row * N + col] = f2bf(acc[a][t][e]);
        else
          ((float*)Cout)[(size_t)row * N + col] = acc[a][t][e];
      }
}

// ---------------------------------------------------------------------------
// Per-head RMSNorm + RoPE, in place on bf16 [B*S][NH*128].
// One wave per (token, head); block = 128 threads = 4 waves.
// ---------------------------------------------------------------------------
__global__ __launch_bounds__(128)
void norm_rope(unsigned short* __restrict__ X, const float* __restrict__ w,
               const float* __restrict__ cosT, const float* __restrict__ sinT,
               int NH, int S_) {
  int wid = blockIdx.x * 4 + (threadIdx.x >> 5);
  int lane = threadIdx.x & 31;
  int head = wid % NH;
  int token = wid / NH;
  int pos = token % S_;
  unsigned short* xp = X + (size_t)token * NH * 128 + head * 128;

  float x[4];
#pragma unroll
  for (int j = 0; j < 4; ++j) x[j] = bf2f(xp[lane + 32 * j]);
  float ss = x[0] * x[0] + x[1] * x[1] + x[2] * x[2] + x[3] * x[3];
#pragma unroll
  for (int m = 16; m >= 1; m >>= 1) ss += __shfl_xor(ss, m, 32);
  float r = rsqrtf(ss * (1.0f / 128.0f) + 1e-6f);

  float y[4];
#pragma unroll
  for (int j = 0; j < 4; ++j) y[j] = x[j] * r * w[lane + 32 * j];

  // rotate_half pairs: (d, d+64)
#pragma unroll
  for (int j = 0; j < 2; ++j) {
    int d = lane + 32 * j;
    float c1 = cosT[pos * 128 + d],      s1 = sinT[pos * 128 + d];
    float c2 = cosT[pos * 128 + d + 64], s2 = sinT[pos * 128 + d + 64];
    float o1 = y[j] * c1 - y[j + 2] * s1;
    float o2 = y[j + 2] * c2 + y[j] * s2;
    xp[d] = f2bf(o1);
    xp[d + 64] = f2bf(o2);
  }
}

// ---------------------------------------------------------------------------
// Causal GQA flash attention, bf16 WMMA with f32 online softmax.
// Block = 256 threads (8 waves); each wave owns 16 query rows; block covers
// 128 query rows of one (b, h). Key loop in 32-wide tiles up to the diagonal.
// K tiles stream in via async global->LDS.
// ---------------------------------------------------------------------------
__global__ __launch_bounds__(256)
void flash_attn(const unsigned short* __restrict__ Q,
                const unsigned short* __restrict__ Km,
                const unsigned short* __restrict__ Vm,
                unsigned short* __restrict__ O) {
  constexpr int S_ = 1024, Hh = 32, KVH = 4, HDc = 128;
  __shared__ unsigned short sK[32 * 128];    // [key][dim]
  __shared__ unsigned short sVt[128 * 32];   // [dim][key]
  __shared__ float sS[8][16 * 32];
  __shared__ unsigned short sP[8][16 * 32];
  __shared__ float sAl[8][16];

  const int tid = threadIdx.x, lane = tid & 31, wave = tid >> 5;
  const int half = lane >> 4, nl = lane & 15;
  const int qt = blockIdx.x;
  const int h = blockIdx.y;
  const int b = blockIdx.z;
  const int kvh = h >> 3;                    // H/K = 8
  const int q0 = qt * 128;
  const int qrow_base = q0 + wave * 16;
  const unsigned sK_base = lds_off(sK);

  // Q fragments resident in registers: 4 chunks over head dim (K=32 each)
  v16bf qf[4];
  {
    const unsigned short* qbase =
        Q + (size_t)(b * S_ + qrow_base) * (Hh * HDc) + h * HDc;
#pragma unroll
    for (int c = 0; c < 4; ++c)
      qf[c] = load_frag_a(qbase + c * 32, Hh * HDc, lane);
  }

  v8f o[8];
#pragma unroll
  for (int t = 0; t < 8; ++t)
#pragma unroll
    for (int e = 0; e < 8; ++e) o[t][e] = 0.0f;
  float m_i = -1e30f, l_i = 0.0f;            // per-row stats in lanes 0..15

  const int nkt = (q0 + 128) / 32;
  const float scale = 0.08838834764831845f;  // 1/sqrt(128)

  for (int jt = 0; jt < nkt; ++jt) {
    const int j0 = jt * 32;
    // K tile (32x128 row-major): async global -> LDS
    int chunk = tid;
#pragma unroll
    for (int it = 0; it < 2; ++it, chunk += 256) {
      int kr = chunk >> 4;
      int dc = (chunk & 15) * 8;
      async_load_b128(
          sK_base + (unsigned)((kr * 128 + dc) * 2),
          Km + (size_t)(b * S_ + j0 + kr) * (KVH * HDc) + kvh * HDc + dc);
    }
    // V tile: load to VGPRs, store transposed [dim][key]
    chunk = tid;
#pragma unroll
    for (int it = 0; it < 2; ++it, chunk += 256) {
      int kr = chunk >> 4;
      int dc = (chunk & 15) * 8;
      uint4v d = *(const uint4v*)(
          Vm + (size_t)(b * S_ + j0 + kr) * (KVH * HDc) + kvh * HDc + dc);
      const unsigned short* ds = (const unsigned short*)&d;
#pragma unroll
      for (int j = 0; j < 8; ++j) sVt[(dc + j) * 32 + kr] = ds[j];
    }
    wait_async0();
    __syncthreads();

    // S = Q * K^T : two 16x16 tiles (keys j0..j0+15, j0+16..j0+31)
    v8f sacc[2];
#pragma unroll
    for (int t = 0; t < 2; ++t) {
#pragma unroll
      for (int e = 0; e < 8; ++e) sacc[t][e] = 0.0f;
#pragma unroll
      for (int c = 0; c < 4; ++c) {
        v16bf kf = load_frag_bt(sK + (t * 16) * 128 + c * 32, 128, lane);
        sacc[t] = __builtin_amdgcn_wmma_f32_16x16x32_bf16(
            false, qf[c], false, kf, (short)0, sacc[t], false, false);
      }
    }
    // scale + causal mask, spill to LDS for row-wise softmax
#pragma unroll
    for (int t = 0; t < 2; ++t)
#pragma unroll
      for (int e = 0; e < 8; ++e) {
        int rr = e + 8 * half;
        int col = j0 + t * 16 + nl;
        int row = qrow_base + rr;
        float v = sacc[t][e] * scale;
        if (col > row) v = -3.0e38f;
        sS[wave][rr * 32 + t * 16 + nl] = v;
      }
    wait_ds0();

    // online softmax: lanes 0..15 each own one query row
    if (lane < 16) {
      float sv[32];
      float mt = -3.0e38f;
#pragma unroll
      for (int j = 0; j < 32; ++j) {
        sv[j] = sS[wave][lane * 32 + j];
        mt = fmaxf(mt, sv[j]);
      }
      float mn = fmaxf(m_i, mt);
      float alpha = __expf(m_i - mn);
      float sum = 0.0f;
#pragma unroll
      for (int j = 0; j < 32; ++j) {
        float p = __expf(sv[j] - mn);
        sum += p;
        sP[wave][lane * 32 + j] = f2bf(p);
      }
      l_i = l_i * alpha + sum;
      m_i = mn;
      sAl[wave][lane] = alpha;
    }
    wait_ds0();

    // rescale running output by alpha(row)
    float al[8];
#pragma unroll
    for (int e = 0; e < 8; ++e) al[e] = sAl[wave][e + 8 * half];
#pragma unroll
    for (int t = 0; t < 8; ++t)
#pragma unroll
      for (int e = 0; e < 8; ++e) o[t][e] *= al[e];

    // O += P * V  (8 WMMAs across head dim)
    v16bf pf = load_frag_a(sP[wave], 32, lane);
#pragma unroll
    for (int t = 0; t < 8; ++t) {
      v16bf vf = load_frag_bt(sVt + (t * 16) * 32, 32, lane);
      o[t] = __builtin_amdgcn_wmma_f32_16x16x32_bf16(
          false, pf, false, vf, (short)0, o[t], false, false);
    }
    __syncthreads();
  }

  // epilogue: divide by l_i, store bf16
  if (lane < 16) sAl[wave][lane] = 1.0f / l_i;
  wait_ds0();
  float il[8];
#pragma unroll
  for (int e = 0; e < 8; ++e) il[e] = sAl[wave][e + 8 * half];
#pragma unroll
  for (int t = 0; t < 8; ++t)
#pragma unroll
    for (int e = 0; e < 8; ++e) {
      int row = b * S_ + qrow_base + e + 8 * half;
      int col = h * HDc + t * 16 + nl;
      O[(size_t)row * (Hh * HDc) + col] = f2bf(o[t][e] * il[e]);
    }
}

// ---------------------------------------------------------------------------
// Launch
// ---------------------------------------------------------------------------
extern "C" void kernel_launch(void* const* d_in, const int* in_sizes, int n_in,
                              void* d_out, int out_size, void* d_ws,
                              size_t ws_size, hipStream_t stream) {
  (void)in_sizes; (void)n_in; (void)out_size; (void)ws_size;
  const float* hs   = (const float*)d_in[0];
  const float* cosT = (const float*)d_in[1];
  const float* sinT = (const float*)d_in[2];
  const float* Wq   = (const float*)d_in[3];
  const float* Wk   = (const float*)d_in[4];
  const float* Wv   = (const float*)d_in[5];
  const float* Wo   = (const float*)d_in[6];
  const float* qw   = (const float*)d_in[7];
  const float* kw   = (const float*)d_in[8];

  const int Bc = 4, Sc = 1024, Dc = 2048, Hc = 32, Kc = 4, HDc = 128;
  const int M = Bc * Sc;        // 4096
  const int NQ = Hc * HDc;      // 4096
  const int NK = Kc * HDc;      // 512

  char* ws = (char*)d_ws;
  size_t off = 0;
  unsigned short* hb  = (unsigned short*)(ws + off); off += (size_t)M * Dc * 2;
  unsigned short* wqb = (unsigned short*)(ws + off); off += (size_t)Dc * NQ * 2;
  unsigned short* wkb = (unsigned short*)(ws + off); off += (size_t)Dc * NK * 2;
  unsigned short* wvb = (unsigned short*)(ws + off); off += (size_t)Dc * NK * 2;
  unsigned short* wob = (unsigned short*)(ws + off); off += (size_t)NQ * Dc * 2;
  unsigned short* Qb  = (unsigned short*)(ws + off); off += (size_t)M * NQ * 2;
  unsigned short* Kb  = (unsigned short*)(ws + off); off += (size_t)M * NK * 2;
  unsigned short* Vb  = (unsigned short*)(ws + off); off += (size_t)M * NK * 2;
  unsigned short* AOb = (unsigned short*)(ws + off);

  cvt_f32_bf16<<<2048, 256, 0, stream>>>(hs, hb, M * Dc);
  cvt_f32_bf16<<<2048, 256, 0, stream>>>(Wq, wqb, Dc * NQ);
  cvt_f32_bf16<<<512, 256, 0, stream>>>(Wk, wkb, Dc * NK);
  cvt_f32_bf16<<<512, 256, 0, stream>>>(Wv, wvb, Dc * NK);
  cvt_f32_bf16<<<2048, 256, 0, stream>>>(Wo, wob, NQ * Dc);

  gemm_bf16_wmma<true><<<dim3(NQ / 128, M / 128), 256, 0, stream>>>(
      hb, wqb, Qb, M, NQ, Dc);
  gemm_bf16_wmma<true><<<dim3(NK / 128, M / 128), 256, 0, stream>>>(
      hb, wkb, Kb, M, NK, Dc);
  gemm_bf16_wmma<true><<<dim3(NK / 128, M / 128), 256, 0, stream>>>(
      hb, wvb, Vb, M, NK, Dc);

  norm_rope<<<(M * Hc) / 4, 128, 0, stream>>>(Qb, qw, cosT, sinT, Hc, Sc);
  norm_rope<<<(M * Kc) / 4, 128, 0, stream>>>(Kb, kw, cosT, sinT, Kc, Sc);

  flash_attn<<<dim3(Sc / 128, Hc, Bc), 256, 0, stream>>>(Qb, Kb, Vb, AOb);

  gemm_bf16_wmma<false><<<dim3(Dc / 128, M / 128), 256, 0, stream>>>(
      AOb, wob, d_out, M, Dc, NQ);
}
